// Merge_81080392614058
// MI455X (gfx1250) — matrix operand order repaired and checked by reading, not compiled, not run on hardware
//
#include <hip/hip_runtime.h>
#include <hip/hip_bf16.h>

// ---------------------------------------------------------------------------
// Problem constants (from reference): B=16, N=4096 tokens, C=256, NPOINT=3072
// ---------------------------------------------------------------------------
#define NB    16      // batches
#define NTOK  4096    // tokens per batch
#define NH    2048    // tokens per half (even/odd split)
#define NC    256     // channels (points)
#define RSEL  1024    // r = N - NPOINT
#define NMRG  3072    // merged token count
#define PADROW 264    // 256 ushorts + 8 pad (16B) -> LDS bank-conflict-free

typedef __bf16 v16bf __attribute__((ext_vector_type(16)));
typedef float  v8f   __attribute__((ext_vector_type(8)));

union FragU { uint4 q[2]; v16bf v; };

// --- CDNA5 async global->LDS staging (ASYNCcnt-tracked), with safe fallback --
#if defined(__AMDGCN__) && __has_builtin(__builtin_amdgcn_global_load_async_to_lds_b128) && __has_builtin(__builtin_amdgcn_s_wait_asynccnt)
#define USE_ASYNC_LDS 1
typedef int v4i_b128 __attribute__((vector_size(16)));
__device__ __forceinline__ __attribute__((address_space(1))) v4i_b128*
as_global(const void* p) {
    return (__attribute__((address_space(1))) v4i_b128*)(unsigned long long)p;
}
__device__ __forceinline__ __attribute__((address_space(3))) v4i_b128*
as_lds(const void* p) {
    // flat LDS addresses carry the LDS offset in the low 32 bits
    return (__attribute__((address_space(3))) v4i_b128*)(unsigned)(unsigned long long)p;
}
#else
#define USE_ASYNC_LDS 0
#endif

__device__ __forceinline__ unsigned short f32_to_bf16(float f) {
    unsigned u = __float_as_uint(f);
    unsigned r = u + 0x7FFFu + ((u >> 16) & 1u);   // round-to-nearest-even
    return (unsigned short)(r >> 16);
}

__device__ __forceinline__ unsigned pack_bf16x2(float lo, float hi) {
    return (unsigned)f32_to_bf16(lo) | ((unsigned)f32_to_bf16(hi) << 16);
}

// ---------------------------------------------------------------------------
// K1: row L2-normalize, split even/odd, pack to bf16.  One wave per row.
// Per-lane contiguous 8-channel chunk -> float4 loads + one b128 store.
// ---------------------------------------------------------------------------
__global__ __launch_bounds__(256)
void normalize_pack_kernel(const float* __restrict__ points,
                           unsigned short* __restrict__ a_bf,
                           unsigned short* __restrict__ b_bf) {
    const int lane = threadIdx.x & 31;
    const int wave = threadIdx.x >> 5;
    const long row = (long)blockIdx.x * 8 + wave;       // 0 .. NB*NTOK-1
    const int b = (int)(row >> 12);                     // /4096
    const int t = (int)(row & 4095);

    const float* p = points + (size_t)row * NC + lane * 8;
    const float4 u0 = *(const float4*)(p);
    const float4 u1 = *(const float4*)(p + 4);

    float s = u0.x*u0.x + u0.y*u0.y + u0.z*u0.z + u0.w*u0.w
            + u1.x*u1.x + u1.y*u1.y + u1.z*u1.z + u1.w*u1.w;
    #pragma unroll
    for (int off = 16; off > 0; off >>= 1)
        s += __shfl_xor(s, off, 32);
    const float inv = rsqrtf(s);

    uint4 o;
    o.x = pack_bf16x2(u0.x * inv, u0.y * inv);
    o.y = pack_bf16x2(u0.z * inv, u0.w * inv);
    o.z = pack_bf16x2(u1.x * inv, u1.y * inv);
    o.w = pack_bf16x2(u1.z * inv, u1.w * inv);

    unsigned short* dst = ((t & 1) ? b_bf : a_bf) +
                          ((size_t)b * NH + (t >> 1)) * NC + lane * 8;
    *(uint4*)dst = o;
}

// ---------------------------------------------------------------------------
// K2: fused  scores = a @ b^T  with row-wise max/argmax, never materialized.
// Block = 256 threads (8 waves); blockIdx.x = t_oct (0..15), blockIdx.y = b.
// Wave w owns t_tile = t_oct*8 + w (16 rows) and sweeps ALL 128 s-tiles.
// Each 16x256 B tile is staged ONCE into LDS (double-buffered, padded rows),
// via GLOBAL_LOAD_ASYNC_TO_LDS when available (ASYNCcnt), else reg-staged.
// All 8 B fragments are pulled from LDS first, then the 8 WMMAs burst.
// C/D layout: VGPR v -> row v + 8*(lane>=16), col = lane&15.
// ---------------------------------------------------------------------------
__global__ __launch_bounds__(256)
void wmma_rowmax_kernel(const unsigned short* __restrict__ a_bf,
                        const unsigned short* __restrict__ b_bf,
                        float* __restrict__ node_max,
                        int* __restrict__ node_idx) {
    const int b    = blockIdx.y;
    const int tid  = threadIdx.x;
    const int lane = tid & 31;
    const int wave = tid >> 5;
    const int t_tile = blockIdx.x * 8 + wave;           // 0..127

    const unsigned short* A = a_bf + (size_t)b * NH * NC;
    const unsigned short* B = b_bf + (size_t)b * NH * NC;

    // ---- load & hold the 16x256 A strip as 8 bf16 fragments (K=32 each) ----
    // ISA 16-bit A 16x32 layout: lanes 0-15 row=lane, K {0..7,16..23};
    // lanes 16-31 row=lane-16, K {8..15,24..31}.
    const int arow = t_tile * 16 + (lane & 15);
    const int akh  = (lane >> 4) * 8;
    v16bf afrag[8];
    #pragma unroll
    for (int k8 = 0; k8 < 8; ++k8) {
        const unsigned short* p = A + (size_t)arow * NC + k8 * 32 + akh;
        FragU f;
        f.q[0] = *(const uint4*)(p);        // K klo .. klo+7
        f.q[1] = *(const uint4*)(p + 16);   // K klo+16 .. klo+23
        afrag[k8] = f.v;
    }

    // ---- LDS double buffer for one 16x256 bf16 B tile ----
    __shared__ __align__(16) unsigned short Btile[2][16][PADROW];

    // cooperative stage mapping: 512 x 16B chunks, 2 per thread
    const int c0 = tid * 2, c1 = tid * 2 + 1;
    const int r0 = c0 >> 5, col0 = (c0 & 31) * 8;
    const int r1 = c1 >> 5, col1 = (c1 & 31) * 8;

#if USE_ASYNC_LDS
    __builtin_amdgcn_global_load_async_to_lds_b128(
        as_global(B + (size_t)r0 * NC + col0), as_lds(&Btile[0][r0][col0]), 0, 0);
    __builtin_amdgcn_global_load_async_to_lds_b128(
        as_global(B + (size_t)r1 * NC + col1), as_lds(&Btile[0][r1][col1]), 0, 0);
    __builtin_amdgcn_s_wait_asynccnt(0);
#else
    uint4 pre0 = *(const uint4*)(B + (size_t)r0 * NC + col0);
    uint4 pre1 = *(const uint4*)(B + (size_t)r1 * NC + col1);
    *(uint4*)&Btile[0][r0][col0] = pre0;
    *(uint4*)&Btile[0][r1][col1] = pre1;
#endif
    __syncthreads();

    float rmax[8];
    int   ridx[8];
    #pragma unroll
    for (int v = 0; v < 8; ++v) { rmax[v] = -3.4e38f; ridx[v] = 0; }

    // B 32x16 layout: col = lane&15; lanes 0-15 K {0..7,16..23}, 16-31 K {8..15,24..31}.
    const int bn  = lane & 15;
    const int bkh = (lane >> 4) * 8;

    for (int s_tile = 0; s_tile < NH / 16; ++s_tile) {
        const int cur = s_tile & 1;
        const bool more = (s_tile + 1) < NH / 16;
        if (more) {   // stage next tile into the other buffer
            const unsigned short* nb = B + (size_t)(s_tile + 1) * 16 * NC;
#if USE_ASYNC_LDS
            __builtin_amdgcn_global_load_async_to_lds_b128(
                as_global(nb + (size_t)r0 * NC + col0),
                as_lds(&Btile[cur ^ 1][r0][col0]), 0, 0);
            __builtin_amdgcn_global_load_async_to_lds_b128(
                as_global(nb + (size_t)r1 * NC + col1),
                as_lds(&Btile[cur ^ 1][r1][col1]), 0, 0);
#else
            pre0 = *(const uint4*)(nb + (size_t)r0 * NC + col0);
            pre1 = *(const uint4*)(nb + (size_t)r1 * NC + col1);
#endif
        }

        // pull ALL 8 B fragments from LDS (one ds clause), then WMMA burst
        FragU bfr[8];
        #pragma unroll
        for (int k8 = 0; k8 < 8; ++k8) {
            const unsigned short* p = &Btile[cur][bn][k8 * 32 + bkh];
            bfr[k8].q[0] = *(const uint4*)(p);
            bfr[k8].q[1] = *(const uint4*)(p + 16);
        }
        v8f acc = {0.f, 0.f, 0.f, 0.f, 0.f, 0.f, 0.f, 0.f};
        #pragma unroll
        for (int k8 = 0; k8 < 8; ++k8) {
            acc = __builtin_amdgcn_wmma_f32_16x16x32_bf16(
                      false, afrag[k8], false, bfr[k8].v, (short)0, acc,
                      false, false);
        }

        const int col = s_tile * 16 + bn;
        #pragma unroll
        for (int v = 0; v < 8; ++v) {
            float val = acc[v];
            if (val > rmax[v]) { rmax[v] = val; ridx[v] = col; }
        }

        if (more) {
#if USE_ASYNC_LDS
            __builtin_amdgcn_s_wait_asynccnt(0);
#else
            *(uint4*)&Btile[cur ^ 1][r0][col0] = pre0;
            *(uint4*)&Btile[cur ^ 1][r1][col1] = pre1;
#endif
        }
        __syncthreads();
    }

    // ---- cross-lane argmax reduce: butterfly within each 16-lane half ----
    #pragma unroll
    for (int v = 0; v < 8; ++v) {
        float val = rmax[v];
        int   idx = ridx[v];
        #pragma unroll
        for (int m = 8; m > 0; m >>= 1) {
            float ov = __shfl_xor(val, m, 32);
            int   oi = __shfl_xor(idx, m, 32);
            if (ov > val || (ov == val && oi < idx)) { val = ov; idx = oi; }
        }
        if ((lane & 15) == 0) {
            const int row = v + ((lane >> 4) * 8);
            node_max[(size_t)b * NH + t_tile * 16 + row] = val;
            node_idx[(size_t)b * NH + t_tile * 16 + row] = idx;
        }
    }
}

// ---------------------------------------------------------------------------
// K3: per-batch bitonic sort (descending score, index tie-break) in LDS,
// emit src/unm/dst index arrays and scatter counts.  1024 thr / 2048 keys.
// ---------------------------------------------------------------------------
__global__ __launch_bounds__(1024)
void sort_select_kernel(const float* __restrict__ node_max,
                        const int* __restrict__ node_idx,
                        int* __restrict__ src_idx,
                        int* __restrict__ unm_idx,
                        int* __restrict__ dst_idx,
                        float* __restrict__ cnt) {
    __shared__ unsigned long long keys[NH];
    const int b = blockIdx.x;
    const int t = threadIdx.x;

    for (int i = t; i < NH; i += 1024) {
        float v = node_max[(size_t)b * NH + i];
        unsigned u = __float_as_uint(v);
        unsigned ord = (u & 0x80000000u) ? ~u : (u | 0x80000000u); // asc order
        keys[i] = ((unsigned long long)(~ord) << 32) | (unsigned)i; // desc + stable
        cnt[(size_t)b * NH + i] = 1.0f;
    }
    __syncthreads();

    for (int k = 2; k <= NH; k <<= 1) {
        for (int j = k >> 1; j > 0; j >>= 1) {
            const int i = ((t / j) * (2 * j)) + (t % j);
            const int p = i + j;
            const bool asc = ((i & k) == 0);
            unsigned long long x = keys[i], y = keys[p];
            if ((x > y) == asc) { keys[i] = y; keys[p] = x; }
            __syncthreads();
        }
    }

    // top r=1024 -> src, rest -> unm
    {
        const unsigned s = (unsigned)(keys[t] & 0xFFFFFFFFull);
        src_idx[(size_t)b * RSEL + t] = (int)s;
        const int d = node_idx[(size_t)b * NH + (int)s];
        dst_idx[(size_t)b * RSEL + t] = d;
        const unsigned um = (unsigned)(keys[RSEL + t] & 0xFFFFFFFFull);
        unm_idx[(size_t)b * RSEL + t] = (int)um;
        atomicAdd(&cnt[(size_t)b * NH + d], 1.0f);   // batch-local
    }
}

// ---------------------------------------------------------------------------
// Row-copy helper: float4 path when C%4==0, scalar otherwise.
// ---------------------------------------------------------------------------
__device__ __forceinline__ void copy_row(float* __restrict__ dst,
                                         const float* __restrict__ src,
                                         int C, int tid, int nthr) {
    if ((C & 3) == 0) {
        const float4* s4 = (const float4*)src;
        float4* d4 = (float4*)dst;
        for (int c = tid; c < (C >> 2); c += nthr) d4[c] = s4[c];
    } else {
        for (int c = tid; c < C; c += nthr) dst[c] = src[c];
    }
}

// K4a: merged base values (unm gather + raw dst copy).  1 block / out token.
__global__ void merge_gather_kernel(const float* __restrict__ x,
                                    const int* __restrict__ unm_idx,
                                    float* __restrict__ merged, int C) {
    const int blk = blockIdx.x;                 // b*NMRG + token
    const int b = blk / NMRG;
    const int tkn = blk - b * NMRG;
    const float* srow;
    if (tkn < RSEL) {
        const int s = unm_idx[(size_t)b * RSEL + tkn];
        srow = x + ((size_t)b * NTOK + 2 * s) * C;        // even token
    } else {
        const int j = tkn - RSEL;
        srow = x + ((size_t)b * NTOK + 2 * j + 1) * C;    // odd token
    }
    float* drow = merged + ((size_t)b * NMRG + tkn) * C;
    copy_row(drow, srow, C, threadIdx.x, blockDim.x);
}

// K4b: scatter-add src tokens into dst region of merged (scalar atomics).
__global__ void scatter_add_kernel(const float* __restrict__ x,
                                   const int* __restrict__ src_idx,
                                   const int* __restrict__ dst_idx,
                                   float* __restrict__ merged, int C) {
    const int blk = blockIdx.x;                 // b*RSEL + i
    const int b = blk >> 10;
    const int i = blk & (RSEL - 1);
    const int s = src_idx[(size_t)b * RSEL + i];
    const int d = dst_idx[(size_t)b * RSEL + i];
    const float* srow = x + ((size_t)b * NTOK + 2 * s) * C;
    float* drow = merged + ((size_t)b * NMRG + RSEL + d) * C;
    for (int c = threadIdx.x; c < C; c += blockDim.x) atomicAdd(&drow[c], srow[c]);
}

// K4c: divide dst region by counts (finalizes merged output).
__global__ void divide_kernel(float* __restrict__ merged,
                              const float* __restrict__ cnt, int C) {
    const int blk = blockIdx.x;                 // b*NH + j
    const int b = blk >> 11;
    const int j = blk & (NH - 1);
    const float inv = 1.0f / cnt[(size_t)b * NH + j];
    float* drow = merged + ((size_t)b * NMRG + RSEL + j) * C;
    if ((C & 3) == 0) {
        float4* d4 = (float4*)drow;
        for (int c = threadIdx.x; c < (C >> 2); c += blockDim.x) {
            float4 v = d4[c];
            v.x *= inv; v.y *= inv; v.z *= inv; v.w *= inv;
            d4[c] = v;
        }
    } else {
        for (int c = threadIdx.x; c < C; c += blockDim.x) drow[c] *= inv;
    }
}

// K5: unmerge back to (B, NTOK, C).  Every slot written exactly once.
__global__ void unmerge_kernel(const float* __restrict__ merged,
                               const int* __restrict__ unm_idx,
                               const int* __restrict__ src_idx,
                               const int* __restrict__ dst_idx,
                               float* __restrict__ out, int C) {
    const int blk = blockIdx.x;                 // b*NTOK + u
    const int b = blk >> 12;
    const int u = blk & (NTOK - 1);
    const float* srow;
    size_t orow;
    if (u < NH) {                                // odd slots <- dst
        srow = merged + ((size_t)b * NMRG + RSEL + u) * C;
        orow = (size_t)b * NTOK + 2 * u + 1;
    } else if (u < NH + RSEL) {                  // even slots <- unm
        const int i = u - NH;
        srow = merged + ((size_t)b * NMRG + i) * C;
        orow = (size_t)b * NTOK + 2 * unm_idx[(size_t)b * RSEL + i];
    } else {                                     // even slots <- dst[dst_idx]
        const int i = u - (NH + RSEL);
        srow = merged + ((size_t)b * NMRG + RSEL + dst_idx[(size_t)b * RSEL + i]) * C;
        orow = (size_t)b * NTOK + 2 * src_idx[(size_t)b * RSEL + i];
    }
    float* drow = out + orow * C;
    copy_row(drow, srow, C, threadIdx.x, blockDim.x);
}

// ---------------------------------------------------------------------------
extern "C" void kernel_launch(void* const* d_in, const int* in_sizes, int n_in,
                              void* d_out, int out_size, void* d_ws, size_t ws_size,
                              hipStream_t stream) {
    const float* points = (const float*)d_in[0];   // (16,4096,256) f32
    const float* xyz    = (const float*)d_in[1];   // (16,4096,3)   f32

    float* out = (float*)d_out;
    const size_t MP = (size_t)NB * NMRG * NC;      // merged_pts
    const size_t CP = (size_t)NB * NTOK * NC;      // compressed_pts
    const size_t MX = (size_t)NB * NMRG * 3;       // merged_xyz
    float* merged_pts = out;
    float* comp_pts   = out + MP;
    float* merged_xyz = comp_pts + CP;
    float* comp_xyz   = merged_xyz + MX;

    // workspace carve (~34 MB)
    char* w = (char*)d_ws;
    unsigned short* a_bf = (unsigned short*)w; w += (size_t)NB * NH * NC * 2;
    unsigned short* b_bf = (unsigned short*)w; w += (size_t)NB * NH * NC * 2;
    float* node_max = (float*)w; w += (size_t)NB * NH * 4;
    int*   node_idx = (int*)w;   w += (size_t)NB * NH * 4;
    int*   src_idx  = (int*)w;   w += (size_t)NB * RSEL * 4;
    int*   unm_idx  = (int*)w;   w += (size_t)NB * RSEL * 4;
    int*   dst_idx  = (int*)w;   w += (size_t)NB * RSEL * 4;
    float* cnt      = (float*)w; w += (size_t)NB * NH * 4;

    normalize_pack_kernel<<<NB * NTOK / 8, 256, 0, stream>>>(points, a_bf, b_bf);

    wmma_rowmax_kernel<<<dim3(16, NB), 256, 0, stream>>>(
        a_bf, b_bf, node_max, node_idx);

    sort_select_kernel<<<NB, 1024, 0, stream>>>(
        node_max, node_idx, src_idx, unm_idx, dst_idx, cnt);

    merge_gather_kernel<<<NB * NMRG, 64, 0, stream>>>(points, unm_idx, merged_pts, NC);
    merge_gather_kernel<<<NB * NMRG, 32, 0, stream>>>(xyz, unm_idx, merged_xyz, 3);

    scatter_add_kernel<<<NB * RSEL, 256, 0, stream>>>(points, src_idx, dst_idx, merged_pts, NC);
    scatter_add_kernel<<<NB * RSEL, 32, 0, stream>>>(xyz, src_idx, dst_idx, merged_xyz, 3);

    divide_kernel<<<NB * NH, 64, 0, stream>>>(merged_pts, cnt, NC);
    divide_kernel<<<NB * NH, 32, 0, stream>>>(merged_xyz, cnt, 3);

    unmerge_kernel<<<NB * NTOK, 64, 0, stream>>>(
        merged_pts, unm_idx, src_idx, dst_idx, comp_pts, NC);
    unmerge_kernel<<<NB * NTOK, 32, 0, stream>>>(
        merged_xyz, unm_idx, src_idx, dst_idx, comp_xyz, 3);
}
